// RWKVEncoder_73220602462464
// MI455X (gfx1250) — matrix operand order repaired and checked by reading, not compiled, not run on hardware
//
#include <hip/hip_runtime.h>
#include <hip/hip_bf16.h>

#define B_ 8
#define T_ 2048
#define C_ 1024
#define M_ (B_*T_)      // 16384 rows in the big GEMMs
#define BC_ (B_*C_)     // 8192 channels
#define NCH 16          // scan chunks
#define LCH (T_/NCH)    // 128 steps per chunk

typedef __attribute__((ext_vector_type(16))) __bf16 bf16x16;
typedef __attribute__((ext_vector_type(8)))  __bf16 bf16x8;
typedef __attribute__((ext_vector_type(8)))  float  f32x8;
typedef __attribute__((ext_vector_type(4)))  unsigned int u32x4;
typedef __attribute__((ext_vector_type(8)))  int i32x8;
typedef __attribute__((ext_vector_type(4)))  int i32x4;

union ABFrag { bf16x16 v; bf16x8 h[2]; };

// ---------------------------------------------------------------------------
// TDM: 2-D tile load Global->LDS via tensor descriptor (cdna5 §8.3-8.6).
// rows x cols (elements, 2 bytes each), row stride = stride_elems.
// Tensor declared == tile so no OOB; tile packed contiguously into LDS.
// ---------------------------------------------------------------------------
__device__ __forceinline__ void tdm_load_2d_bf16(unsigned lds_addr, const void* gptr,
                                                 unsigned rows, unsigned cols,
                                                 unsigned stride_elems) {
  unsigned long long ga = (unsigned long long)gptr;
  u32x4 g0;
  g0[0] = 1u;                                             // count=1 (valid), user mode
  g0[1] = lds_addr;                                       // lds_addr [63:32]
  g0[2] = (unsigned)(ga & 0xFFFFFFFFu);                   // global_addr [95:64]
  g0[3] = (unsigned)((ga >> 32) & 0x1FFFFFFu)             // global_addr [120:96]
        | (2u << 30);                                     // type=2 (image)
  i32x8 g1;
  g1[0] = (int)(1u << 16);                                // data_size=1 -> 2 bytes
  g1[1] = (int)((cols & 0xFFFFu) << 16);                  // tensor_dim0[15:0]
  g1[2] = (int)(((cols >> 16) & 0xFFFFu) |
                ((rows & 0xFFFFu) << 16));                // tensor_dim0 hi | tensor_dim1 lo
  g1[3] = (int)(((rows >> 16) & 0xFFFFu) |
                ((cols & 0xFFFFu) << 16));                // tensor_dim1 hi | tile_dim0
  g1[4] = (int)(rows & 0xFFFFu);                          // tile_dim1 (tile_dim2=0)
  g1[5] = (int)stride_elems;                              // tensor_dim0_stride lo32
  g1[6] = 0;                                              // stride hi | dim1_stride lo
  g1[7] = 0;
  i32x4 z4 = {0, 0, 0, 0};
#if defined(__clang_major__) && (__clang_major__ >= 23)
  i32x8 z8 = {0, 0, 0, 0, 0, 0, 0, 0};
  __builtin_amdgcn_tensor_load_to_lds(g0, g1, z4, z4, z8, 0);
#else
  __builtin_amdgcn_tensor_load_to_lds(g0, g1, z4, z4, 0);
#endif
}

// ---------------------------------------------------------------------------
// 1) Embedding gather + time-mix -> bf16 activations (xk, xv, xr)
// ---------------------------------------------------------------------------
__global__ __launch_bounds__(256)
void embed_mix_kernel(const int* __restrict__ tok,
                      const float* __restrict__ xx_init,
                      const float* __restrict__ emb,
                      const float* __restrict__ tmk,
                      const float* __restrict__ tmv,
                      const float* __restrict__ tmr,
                      __bf16* __restrict__ xk,
                      __bf16* __restrict__ xv,
                      __bf16* __restrict__ xr) {
  int i  = blockIdx.x * blockDim.x + threadIdx.x;   // over B*T*C
  int c  = i & (C_ - 1);
  int bt = i >> 10;
  int t  = bt & (T_ - 1);
  int b  = bt >> 11;
  int tk = tok[bt];
  float x  = emb[(size_t)tk * C_ + c];
  float xp = (t == 0) ? xx_init[b * C_ + c]
                      : emb[(size_t)tok[bt - 1] * C_ + c];
  float mk = tmk[c], mv = tmv[c], mr = tmr[c];
  xk[i] = (__bf16)(x * mk + xp * (1.f - mk));
  xv[i] = (__bf16)(x * mv + xp * (1.f - mv));
  xr[i] = (__bf16)(x * mr + xp * (1.f - mr));
}

// ---------------------------------------------------------------------------
// 2) Weight fp32 -> bf16 (z selects Wk/Wv/Wr)
// ---------------------------------------------------------------------------
__global__ __launch_bounds__(256)
void wcast_kernel(const float* __restrict__ w0, const float* __restrict__ w1,
                  const float* __restrict__ w2,
                  __bf16* __restrict__ o0, __bf16* __restrict__ o1,
                  __bf16* __restrict__ o2) {
  int i = blockIdx.x * blockDim.x + threadIdx.x;
  int z = blockIdx.z;
  const float* s = (z == 0) ? w0 : (z == 1) ? w1 : w2;
  __bf16*      d = (z == 0) ? o0 : (z == 1) ? o1 : o2;
  d[i] = (__bf16)s[i];
}

// ---------------------------------------------------------------------------
// 3) bf16 WMMA GEMM with TDM double-buffered tiles.
//    Block tile 64(M) x 256(N), K-step 32. 8 waves as 2x4; wave tile 32x64
//    = 2x4 WMMA 16x16 tiles (8 wmma / 12 ds_load_b128 per K-step per wave).
//    z selects (xk,Wk,k) / (xv,Wv,v) / (xr,Wr,r+sigmoid).
// ---------------------------------------------------------------------------
#define BM 64
#define BN 256
#define BK 32

__global__ __launch_bounds__(256)
void gemm_bf16_kernel(const __bf16* __restrict__ A0, const __bf16* __restrict__ A1,
                      const __bf16* __restrict__ A2,
                      const __bf16* __restrict__ W0, const __bf16* __restrict__ W1,
                      const __bf16* __restrict__ W2,
                      float* __restrict__ O0, float* __restrict__ O1,
                      float* __restrict__ O2) {
  __shared__ __align__(16) __bf16 As[2][BM][BK];   //  8 KB
  __shared__ __align__(16) __bf16 Bs[2][BN][BK];   // 32 KB

  int z = blockIdx.z;
  const __bf16* A = (z == 0) ? A0 : (z == 1) ? A1 : A2;
  const __bf16* W = (z == 0) ? W0 : (z == 1) ? W1 : W2;
  float*        O = (z == 0) ? O0 : (z == 1) ? O1 : O2;
  const bool do_sigmoid = (z == 2);

  const int n0 = blockIdx.x * BN;
  const int m0 = blockIdx.y * BM;

  const int tid  = threadIdx.x;
  const int wave = tid >> 5;        // 0..7 (wave32)
  const int lane = tid & 31;
  const int wm   = wave >> 2;       // 0..1 : 32-row slice
  const int wn   = wave & 3;        // 0..3 : 64-col slice
  const int half = lane >> 4;
  const int l16  = lane & 15;

  f32x8 acc[2][4] = {};

  const int NSTEP = C_ / BK;

  if (tid < 32) {  // wave 0 drives the Tensor Data Mover
    tdm_load_2d_bf16((unsigned)(size_t)&As[0][0][0], A + (size_t)m0 * C_, BM, BK, C_);
    tdm_load_2d_bf16((unsigned)(size_t)&Bs[0][0][0], W + (size_t)n0 * C_, BN, BK, C_);
  }

  for (int s = 0; s < NSTEP; ++s) {
    const int buf = s & 1;
    if (s + 1 < NSTEP) {
      if (tid < 32) {
        tdm_load_2d_bf16((unsigned)(size_t)&As[buf ^ 1][0][0],
                         A + (size_t)m0 * C_ + (s + 1) * BK, BM, BK, C_);
        tdm_load_2d_bf16((unsigned)(size_t)&Bs[buf ^ 1][0][0],
                         W + (size_t)n0 * C_ + (s + 1) * BK, BN, BK, C_);
        // leave next-step's 2 DMAs in flight (per-wave in-order completion)
        __builtin_amdgcn_s_wait_tensorcnt(2);
      }
    } else {
      if (tid < 32) __builtin_amdgcn_s_wait_tensorcnt(0);
    }
    __syncthreads();

    // Fragment loads per documented 16-bit WMMA VGPR layouts:
    // A(16x32): lanes 0-15 -> K 0..7 & 16..23, lanes 16-31 -> K 8..15 & 24..31
    // B(32x16): lanes 0-15 -> K 0..7 & 16..23, lanes 16-31 -> K 8..15 & 24..31 (lane%16 = N)
    ABFrag afrag[2], bfrag[4];
#pragma unroll
    for (int i = 0; i < 2; ++i) {
      int ar = wm * 32 + i * 16 + l16;
      afrag[i].h[0] = *(const bf16x8*)&As[buf][ar][half * 8];
      afrag[i].h[1] = *(const bf16x8*)&As[buf][ar][16 + half * 8];
    }
#pragma unroll
    for (int j = 0; j < 4; ++j) {
      int br = wn * 64 + j * 16 + l16;
      bfrag[j].h[0] = *(const bf16x8*)&Bs[buf][br][half * 8];
      bfrag[j].h[1] = *(const bf16x8*)&Bs[buf][br][16 + half * 8];
    }
#pragma unroll
    for (int i = 0; i < 2; ++i)
#pragma unroll
      for (int j = 0; j < 4; ++j)
        acc[i][j] = __builtin_amdgcn_wmma_f32_16x16x32_bf16(
            false, afrag[i].v, false, bfrag[j].v,
            (short)0, acc[i][j], false, false);
    __syncthreads();   // keep buf stable until everyone is done reading it
  }

  // C/D layout: VGPR e -> row (e + half*8), lane%16 -> col
#pragma unroll
  for (int i = 0; i < 2; ++i) {
    int mbase = m0 + wm * 32 + i * 16 + half * 8;
#pragma unroll
    for (int j = 0; j < 4; ++j) {
      int n = n0 + wn * 64 + j * 16 + l16;
#pragma unroll
      for (int e = 0; e < 8; ++e) {
        float val = acc[i][j][e];
        if (do_sigmoid) val = 1.f / (1.f + __expf(-val));
        O[(size_t)(mbase + e) * C_ + n] = val;
      }
    }
  }
}

// ---------------------------------------------------------------------------
// 4a) Chunk summaries: run the (aa,bb,pp) state update over each length-128
//     chunk starting from the zero state (aa=bb=0, pp=-inf).
// ---------------------------------------------------------------------------
__global__ __launch_bounds__(256)
void wkv_summary_kernel(const float* __restrict__ k, const float* __restrict__ v,
                        const float* __restrict__ tdec,
                        float* __restrict__ sa, float* __restrict__ sb,
                        float* __restrict__ sp) {
  int g = blockIdx.x * blockDim.x + threadIdx.x;  // over NCH*BC
  int idx = g & (BC_ - 1);
  int j = g >> 13;
  int c = idx & (C_ - 1);
  int b = idx >> 10;
  float w = -__expf(tdec[c]);
  float aa = 0.f, bb = 0.f, pp = -1e30f;
  const float* kp = k + ((size_t)b * T_ + j * LCH) * C_ + c;
  const float* vp = v + ((size_t)b * T_ + j * LCH) * C_ + c;
  for (int t = 0; t < LCH; ++t) {
    float kt = kp[(size_t)t * C_];
    float vt = vp[(size_t)t * C_];
    float ww2 = pp + w;
    float p2  = fmaxf(ww2, kt);
    float e1b = __expf(ww2 - p2);
    float e2b = __expf(kt - p2);
    aa = e1b * aa + e2b * vt;
    bb = e1b * bb + e2b;
    pp = p2;
  }
  sa[g] = aa; sb[g] = bb; sp[g] = pp;
}

// ---------------------------------------------------------------------------
// 4b) Sequential combine over the 16 chunks (per channel): the chunk map is
//     state_out = e^{w*L} * state_in (+) summary  -- linear, constant decay.
//     Records each chunk's ENTRY state.
// ---------------------------------------------------------------------------
__global__ __launch_bounds__(256)
void wkv_combine_kernel(const float* __restrict__ aa0, const float* __restrict__ bb0,
                        const float* __restrict__ pp0, const float* __restrict__ tdec,
                        const float* __restrict__ sa, const float* __restrict__ sb,
                        const float* __restrict__ sp,
                        float* __restrict__ ia, float* __restrict__ ib,
                        float* __restrict__ ip) {
  int idx = blockIdx.x * blockDim.x + threadIdx.x;  // over BC
  int c = idx & (C_ - 1);
  float wL = -__expf(tdec[c]) * (float)LCH;
  float aa = aa0[idx], bb = bb0[idx], pp = pp0[idx];
  for (int j = 0; j < NCH; ++j) {
    int g = j * BC_ + idx;
    ia[g] = aa; ib[g] = bb; ip[g] = pp;
    float pd = pp + wL;
    float spj = sp[g];
    float q = fmaxf(pd, spj);
    float e1 = __expf(pd - q);
    float e2 = __expf(spj - q);
    aa = e1 * aa + e2 * sa[g];
    bb = e1 * bb + e2 * sb[g];
    pp = q;
  }
}

// ---------------------------------------------------------------------------
// 4c) Replay each chunk (now in parallel) with its correct entry state,
//     accumulating partial sums of r*wkv; last chunk records the final value.
// ---------------------------------------------------------------------------
__global__ __launch_bounds__(256)
void wkv_chunk_scan_kernel(const float* __restrict__ k, const float* __restrict__ v,
                           const float* __restrict__ r,
                           const float* __restrict__ ia, const float* __restrict__ ib,
                           const float* __restrict__ ip,
                           const float* __restrict__ tdec, const float* __restrict__ tfirst,
                           float* __restrict__ psum, float* __restrict__ s1) {
  int g = blockIdx.x * blockDim.x + threadIdx.x;  // over NCH*BC
  int idx = g & (BC_ - 1);
  int j = g >> 13;
  int c = idx & (C_ - 1);
  int b = idx >> 10;
  float w = -__expf(tdec[c]);
  float u = tfirst[c];
  float aa = ia[g], bb = ib[g], pp = ip[g];
  const float* kp = k + ((size_t)b * T_ + j * LCH) * C_ + c;
  const float* vp = v + ((size_t)b * T_ + j * LCH) * C_ + c;
  const float* rp = r + ((size_t)b * T_ + j * LCH) * C_ + c;
  float sum = 0.f, last = 0.f;
  for (int t = 0; t < LCH; ++t) {
    float kt = kp[(size_t)t * C_];
    float vt = vp[(size_t)t * C_];
    float rt = rp[(size_t)t * C_];
    float ww = u + kt;
    float p  = fmaxf(pp, ww);
    float e1 = __expf(pp - p);
    float e2 = __expf(ww - p);
    float wkv = (e1 * aa + e2 * vt) / (e1 * bb + e2);
    float rw = rt * wkv;
    sum += rw;
    last = rw;
    float ww2 = pp + w;
    float p2  = fmaxf(ww2, kt);
    float e1b = __expf(ww2 - p2);
    float e2b = __expf(kt - p2);
    aa = e1b * aa + e2b * vt;
    bb = e1b * bb + e2b;
    pp = p2;
  }
  psum[g] = sum;
  if (j == NCH - 1) s1[idx] = last;
}

// ---------------------------------------------------------------------------
// 4d) Reduce chunk partial sums -> s2 (mean over T)
// ---------------------------------------------------------------------------
__global__ __launch_bounds__(256)
void reduce_psum_kernel(const float* __restrict__ psum, float* __restrict__ s2) {
  int idx = blockIdx.x * blockDim.x + threadIdx.x;  // over BC
  float s = 0.f;
#pragma unroll
  for (int j = 0; j < NCH; ++j) s += psum[j * BC_ + idx];
  s2[idx] = s * (1.0f / T_);
}

// ---------------------------------------------------------------------------
// 5) hx = 0.5 * (s1 + s2) @ Wo^T ; out = stack([hx, hx])
// ---------------------------------------------------------------------------
__global__ __launch_bounds__(256)
void out_kernel(const float* __restrict__ s1, const float* __restrict__ s2,
                const float* __restrict__ Wo, float* __restrict__ out) {
  int d = blockIdx.x * blockDim.x + threadIdx.x;
  int b = blockIdx.y;
  const float* ss1 = s1 + b * C_;
  const float* ss2 = s2 + b * C_;
  const float* wrow = Wo + (size_t)d * C_;
  float acc = 0.f;
  for (int c = 0; c < C_; ++c) acc += (ss1[c] + ss2[c]) * wrow[c];
  float hx = 0.5f * acc;
  out[b * C_ + d] = hx;
  out[B_ * C_ + b * C_ + d] = hx;
}

// ---------------------------------------------------------------------------
extern "C" void kernel_launch(void* const* d_in, const int* in_sizes, int n_in,
                              void* d_out, int out_size, void* d_ws, size_t ws_size,
                              hipStream_t stream) {
  const int*   tok     = (const int*)d_in[0];
  const float* xx_init = (const float*)d_in[1];
  const float* aa_init = (const float*)d_in[2];
  const float* bb_init = (const float*)d_in[3];
  const float* pp_init = (const float*)d_in[4];
  const float* emb     = (const float*)d_in[5];
  const float* tmk     = (const float*)d_in[6];
  const float* tmv     = (const float*)d_in[7];
  const float* tmr     = (const float*)d_in[8];
  const float* tdec    = (const float*)d_in[9];
  const float* tfirst  = (const float*)d_in[10];
  const float* Wk      = (const float*)d_in[11];
  const float* Wv      = (const float*)d_in[12];
  const float* Wr      = (const float*)d_in[13];
  const float* Wo      = (const float*)d_in[14];
  float* out = (float*)d_out;

  char* ws = (char*)d_ws;
  size_t off = 0;
  auto alloc = [&](size_t bytes) -> char* {
    char* p = ws + off;
    off += (bytes + 255) & ~(size_t)255;
    return p;
  };
  const size_t szA = (size_t)M_ * C_ * sizeof(__bf16);   // 32 MB
  const size_t szW = (size_t)C_ * C_ * sizeof(__bf16);   //  2 MB
  const size_t szO = (size_t)M_ * C_ * sizeof(float);    // 64 MB
  const size_t szS = (size_t)BC_ * sizeof(float);        // 32 KB
  const size_t szC = (size_t)NCH * BC_ * sizeof(float);  // 512 KB

  __bf16* xk  = (__bf16*)alloc(szA);
  __bf16* xv  = (__bf16*)alloc(szA);
  __bf16* xr  = (__bf16*)alloc(szA);
  __bf16* wkb = (__bf16*)alloc(szW);
  __bf16* wvb = (__bf16*)alloc(szW);
  __bf16* wrb = (__bf16*)alloc(szW);
  float*  kb  = (float*)alloc(szO);
  float*  vb  = (float*)alloc(szO);
  float*  rb  = (float*)alloc(szO);
  float*  sa  = (float*)alloc(szC);
  float*  sb  = (float*)alloc(szC);
  float*  sp  = (float*)alloc(szC);
  float*  iaa = (float*)alloc(szC);
  float*  ibb = (float*)alloc(szC);
  float*  ipp = (float*)alloc(szC);
  float*  ps  = (float*)alloc(szC);
  float*  s1  = (float*)alloc(szS);
  float*  s2  = (float*)alloc(szS);
  (void)in_sizes; (void)n_in; (void)out_size; (void)ws_size;

  embed_mix_kernel<<<(M_ * C_) / 256, 256, 0, stream>>>(
      tok, xx_init, emb, tmk, tmv, tmr, xk, xv, xr);

  wcast_kernel<<<dim3((C_ * C_) / 256, 1, 3), 256, 0, stream>>>(
      Wk, Wv, Wr, wkb, wvb, wrb);

  gemm_bf16_kernel<<<dim3(C_ / BN, M_ / BM, 3), 256, 0, stream>>>(
      xk, xv, xr, wkb, wvb, wrb, kb, vb, rb);

  wkv_summary_kernel<<<(NCH * BC_) / 256, 256, 0, stream>>>(
      kb, vb, tdec, sa, sb, sp);

  wkv_combine_kernel<<<BC_ / 256, 256, 0, stream>>>(
      aa_init, bb_init, pp_init, tdec, sa, sb, sp, iaa, ibb, ipp);

  wkv_chunk_scan_kernel<<<(NCH * BC_) / 256, 256, 0, stream>>>(
      kb, vb, rb, iaa, ibb, ipp, tdec, tfirst, ps, s1);

  reduce_psum_kernel<<<BC_ / 256, 256, 0, stream>>>(ps, s2);

  out_kernel<<<dim3(C_ / 256, B_), 256, 0, stream>>>(s1, s2, Wo, out);
}
